// NMRGraphEncoder_32633161515657
// MI455X (gfx1250) — compile-verified, matrix operand order checked by reading, not run on hardware
//
#include <hip/hip_runtime.h>
#include <hip/hip_bf16.h>
#include <math.h>

// Problem constants (match reference)
#define NN 50000      // nodes  (divisible by 16 -> 3125 tiles)
#define EE 800000     // edges  (divisible by 8)
#define GG 512        // graphs
#define DD 128        // channels (HEADS*HID)
#define HIDN 64

typedef __attribute__((ext_vector_type(16))) _Float16 v16h;
typedef __attribute__((ext_vector_type(8)))  _Float16 v8h;
typedef __attribute__((ext_vector_type(8)))  float    v8f;
typedef __attribute__((ext_vector_type(4)))  float    f4;

// Packed-weight chunk size per matrix: 8 tiles * 4 ksteps * 32 lanes * 16 f16
#define PWCHUNK 16384

// ---------------------------------------------------------------------------
// Weight pre-pack: W [128out x 128in] f32 -> per-lane WMMA B-fragment f16
// layout, so the GEMM hot loop is pure global_load_b128 + v_wmma.
// t = (jj*4 + kb4)*32 + lane ; each thread emits 16 contiguous f16 (32 B).
// B-frag (16-bit, 32x16): lane n = lane&15, half = lane>>4 holds
//   k = kb + half*8 + [0..7]  and  k = kb + 16 + half*8 + [0..7]
// ---------------------------------------------------------------------------
__global__ __launch_bounds__(256)
void pack_w(const float* __restrict__ W, _Float16* __restrict__ pw)
{
    const int t    = blockIdx.x * blockDim.x + threadIdx.x;   // 0..1023
    const int lane = t & 31;
    const int kb   = ((t >> 5) & 3) * 32;
    const int jj   = t >> 7;                                  // 0..7
    const int half = lane >> 4;
    const int n    = (jj << 4) + (lane & 15);
    const float* wrow = W + (size_t)n * DD + kb + half * 8;
    _Float16* d = pw + ((size_t)t << 4);
    #pragma unroll
    for (int i = 0; i < 8; ++i) {
        d[i]     = (_Float16)wrow[i];
        d[i + 8] = (_Float16)wrow[16 + i];
    }
}

// ---------------------------------------------------------------------------
// Fused node-linear kernel: [Q|K|V|Skip] = X @ W^T + b  using WMMA f16->f32.
// Block = 256 threads (8 waves), one 16-row node tile per block. X tile is
// converted to f16 during the cooperative LDS fill; each wave hoists its 4
// A-fragments (shared across all 4 output matrices) and then runs
// 4 matrices x 4 ksteps of {2x global_load_b128 + v_wmma}.
// ---------------------------------------------------------------------------
__device__ __forceinline__ void wmma_tile(const v16h A[4],
                                          const _Float16* __restrict__ pw,
                                          int jj, int lane,
                                          const float* __restrict__ bias,
                                          float* __restrict__ O,
                                          int rowbase, int half, int mn)
{
    v8f acc = {};
    #pragma unroll
    for (int kb4 = 0; kb4 < 4; ++kb4) {
        const v8h* pb = (const v8h*)(pw + ((size_t)((jj * 4 + kb4) * 32 + lane) << 4));
        const v8h blo = pb[0];
        const v8h bhi = pb[1];
        const v16h b = __builtin_shufflevector(blo, bhi,
                           0, 1, 2, 3, 4, 5, 6, 7, 8, 9, 10, 11, 12, 13, 14, 15);
        acc = __builtin_amdgcn_wmma_f32_16x16x32_f16(
                  false, A[kb4], false, b, (short)0, acc, false, false);
    }
    const int ch = (jj << 4) + mn;
    const float bv = bias[ch];
    const int mb = half * 8;
    #pragma unroll
    for (int r = 0; r < 8; ++r)
        O[(size_t)(rowbase + mb + r) * DD + ch] = acc[r] + bv;
}

__global__ __launch_bounds__(256)
void gemm_qkvs(const float* __restrict__ X, const _Float16* __restrict__ PW,
               const float* __restrict__ bq, const float* __restrict__ bk,
               const float* __restrict__ bv, const float* __restrict__ bs,
               float* __restrict__ Q, float* __restrict__ K,
               float* __restrict__ V, float* __restrict__ S)
{
    __shared__ _Float16 xs[16][136];   // 136-f16 pitch: 16B rows, conflict-free
    const int tid = threadIdx.x;
    const int rowbase = blockIdx.x * 16;

    // Cooperative load + f32->f16 convert of the 16x128 X tile (8 elem/thread).
    {
        const int idx = tid * 8;
        const int r = idx >> 7;
        const int c = idx & 127;
        const float4* src = (const float4*)(X + (size_t)(rowbase + r) * DD + c);
        const float4 a0 = src[0];
        const float4 a1 = src[1];
        v8h hv;
        hv[0] = (_Float16)a0.x; hv[1] = (_Float16)a0.y;
        hv[2] = (_Float16)a0.z; hv[3] = (_Float16)a0.w;
        hv[4] = (_Float16)a1.x; hv[5] = (_Float16)a1.y;
        hv[6] = (_Float16)a1.z; hv[7] = (_Float16)a1.w;
        *(v8h*)&xs[r][c] = hv;
    }
    __syncthreads();

    const int wave = tid >> 5;
    const int lane = tid & 31;
    const int half = lane >> 4;
    const int mn   = lane & 15;

    // Hoisted A fragments: shared by all four output matrices.
    v16h A[4];
    #pragma unroll
    for (int kb4 = 0; kb4 < 4; ++kb4) {
        const v8h alo = *(const v8h*)&xs[mn][kb4 * 32 + half * 8];
        const v8h ahi = *(const v8h*)&xs[mn][kb4 * 32 + 16 + half * 8];
        A[kb4] = __builtin_shufflevector(alo, ahi,
                     0, 1, 2, 3, 4, 5, 6, 7, 8, 9, 10, 11, 12, 13, 14, 15);
    }

    // Concrete pointers per matrix -> global_load path (no flat selects).
    wmma_tile(A, PW,               wave, lane, bq, Q, rowbase, half, mn);
    wmma_tile(A, PW + PWCHUNK,     wave, lane, bk, K, rowbase, half, mn);
    wmma_tile(A, PW + 2 * PWCHUNK, wave, lane, bv, V, rowbase, half, mn);
    wmma_tile(A, PW + 3 * PWCHUNK, wave, lane, bs, S, rowbase, half, mn);
}

// ---------------------------------------------------------------------------
// Edge attention logits: one wave32 per edge. Lane l covers dims [4l,4l+4).
// Lanes 0-15 -> head 0, lanes 16-31 -> head 1. In-wave shfl_xor reduction.
// Segment-max via monotonic uint key atomicMax (IEEE order-preserving map).
// ---------------------------------------------------------------------------
__global__ __launch_bounds__(256)
void edge_alpha(const int* __restrict__ ei, const float* __restrict__ Q,
                const float* __restrict__ K, const float* __restrict__ ea,
                const float* __restrict__ We,
                float* __restrict__ alpha, unsigned int* __restrict__ amax)
{
    const int e    = (blockIdx.x << 3) + (threadIdx.x >> 5);
    const int lane = threadIdx.x & 31;
    const int src  = ei[e];
    const int dst  = ei[EE + e];
    const float eav = ea[e];
    const int d = lane * 4;

    f4 q  = *(const f4*)(Q  + (size_t)dst * DD + d);
    f4 k  = *(const f4*)(K  + (size_t)src * DD + d);
    f4 w  = *(const f4*)(We + d);
    f4 kj = k + eav * w;
    float p = q.x * kj.x + q.y * kj.y + q.z * kj.z + q.w * kj.w;

    p += __shfl_xor(p, 1, 32);
    p += __shfl_xor(p, 2, 32);
    p += __shfl_xor(p, 4, 32);
    p += __shfl_xor(p, 8, 32);

    if ((lane & 15) == 0) {
        const int h = lane >> 4;
        const float al = p * 0.125f;               // 1/sqrt(64)
        alpha[(size_t)e * 2 + h] = al;
        const unsigned int fb  = __float_as_uint(al);
        const unsigned int key = (fb & 0x80000000u) ? ~fb : (fb | 0x80000000u);
        atomicMax(&amax[(size_t)dst * 2 + h], key);
    }
}

// Decode amax keys back to float; non-finite (no incoming edges) -> 0.
__global__ void amax_fix(float* __restrict__ amax)
{
    const int i = blockIdx.x * blockDim.x + threadIdx.x;
    if (i >= NN * 2) return;
    const unsigned int key = __float_as_uint(amax[i]);
    const unsigned int fb  = (key & 0x80000000u) ? (key & 0x7FFFFFFFu) : ~key;
    const float f = __uint_as_float(fb);
    amax[i] = __builtin_isfinite(f) ? f : 0.0f;
}

// ex = exp(alpha - amax[dst]); accumulate softmax denominators.
__global__ void edge_exp(const int* __restrict__ ei, float* __restrict__ alpha,
                         const float* __restrict__ amax, float* __restrict__ den)
{
    const int t = blockIdx.x * blockDim.x + threadIdx.x;   // e*2 + h
    if (t >= EE * 2) return;
    const int e = t >> 1, h = t & 1;
    const int dst = ei[EE + e];
    const float ex = __expf(alpha[t] - amax[(size_t)dst * 2 + h]);
    alpha[t] = ex;                                          // in-place: alpha -> ex
    atomicAdd(&den[(size_t)dst * 2 + h], ex);
}

// msg = (V[src] + e) * a, scatter-added into the skip-initialized accumulator.
__global__ __launch_bounds__(256)
void edge_msg(const int* __restrict__ ei, const float* __restrict__ V,
              const float* __restrict__ ea, const float* __restrict__ We,
              const float* __restrict__ ex, const float* __restrict__ den,
              float* __restrict__ out)
{
    const int e    = (blockIdx.x << 3) + (threadIdx.x >> 5);
    const int lane = threadIdx.x & 31;
    const int src  = ei[e];
    const int dst  = ei[EE + e];
    const int h    = lane >> 4;
    const float a  = ex[(size_t)e * 2 + h] / (den[(size_t)dst * 2 + h] + 1e-16f);
    const float eav = ea[e];
    const int d = lane * 4;

    f4 v = *(const f4*)(V  + (size_t)src * DD + d);
    f4 w = *(const f4*)(We + d);
    f4 m = (v + eav * w) * a;
    float* o = out + (size_t)dst * DD + d;
    atomicAdd(o + 0, m.x);
    atomicAdd(o + 1, m.y);
    atomicAdd(o + 2, m.z);
    atomicAdd(o + 3, m.w);
}

__global__ void copy_act(const float* __restrict__ in, float* __restrict__ outp,
                         int n, int doRelu)
{
    const int i = blockIdx.x * blockDim.x + threadIdx.x;
    if (i < n) {
        const float v = in[i];
        outp[i] = doRelu ? fmaxf(v, 0.0f) : v;
    }
}

// Mean-pool accumulation over sorted batch (atomics into [G,128] + counts).
__global__ void pool_sum(const float* __restrict__ h, const int* __restrict__ batch,
                         float* __restrict__ gsum)
{
    const int t = blockIdx.x * blockDim.x + threadIdx.x;   // n*128 + d
    if (t >= NN * DD) return;
    const int n = t >> 7, d = t & 127;
    atomicAdd(&gsum[(size_t)batch[n] * DD + d], h[t]);
}

__global__ void pool_cnt(const int* __restrict__ batch, float* __restrict__ gcnt)
{
    const int n = blockIdx.x * blockDim.x + threadIdx.x;
    if (n < NN) atomicAdd(&gcnt[batch[n]], 1.0f);
}

// Fused per-graph MLP: mean -> 64 relu -> 32 relu -> 16. One block per graph.
__global__ __launch_bounds__(128)
void graph_mlp(const float* __restrict__ gsum, const float* __restrict__ gcnt,
               const float* __restrict__ W1, const float* __restrict__ b1,
               const float* __restrict__ W2, const float* __restrict__ b2,
               const float* __restrict__ W3, const float* __restrict__ b3,
               float* __restrict__ outp)
{
    __shared__ float g[DD];
    __shared__ float h1[HIDN];
    __shared__ float h2[HIDN / 2];
    const int gi = blockIdx.x, t = threadIdx.x;
    const float cnt = fmaxf(gcnt[gi], 1.0f);
    g[t] = gsum[(size_t)gi * DD + t] / cnt;
    __syncthreads();
    if (t < HIDN) {
        float s = b1[t];
        for (int k = 0; k < DD; ++k) s += W1[t * DD + k] * g[k];
        h1[t] = fmaxf(s, 0.0f);
    }
    __syncthreads();
    if (t < HIDN / 2) {
        float s = b2[t];
        for (int k = 0; k < HIDN; ++k) s += W2[t * HIDN + k] * h1[k];
        h2[t] = fmaxf(s, 0.0f);
    }
    __syncthreads();
    if (t < HIDN / 4) {
        float s = b3[t];
        for (int k = 0; k < HIDN / 2; ++k) s += W3[t * (HIDN / 2) + k] * h2[k];
        outp[(size_t)gi * (HIDN / 4) + t] = s;
    }
}

// ---------------------------------------------------------------------------
extern "C" void kernel_launch(void* const* d_in, const int* in_sizes, int n_in,
                              void* d_out, int out_size, void* d_ws, size_t ws_size,
                              hipStream_t stream)
{
    (void)in_sizes; (void)n_in; (void)out_size; (void)ws_size;

    const float* x     = (const float*)d_in[0];
    const int*   ei    = (const int*)d_in[1];
    const int*   batch = (const int*)d_in[2];
    const float* ea    = (const float*)d_in[3];
    // Per-layer params start at 4 and 13: Wq,bq,Wk,bk,Wv,bv,We,Wskip,bskip
    const float* W1 = (const float*)d_in[22];
    const float* b1 = (const float*)d_in[23];
    const float* W2 = (const float*)d_in[24];
    const float* b2 = (const float*)d_in[25];
    const float* W3 = (const float*)d_in[26];
    const float* b3 = (const float*)d_in[27];

    float* out = (float*)d_out;

    // Workspace layout (floats).  Total ~34M floats (~136 MB).
    const size_t NQ = (size_t)NN * DD;
    float* Q   = (float*)d_ws;
    float* K   = Q + NQ;
    float* V   = K + NQ;
    float* S   = V + NQ;              // skip + attention accumulator
    float* H   = S + NQ;              // layer-0 output / layer-1 input
    float* AL  = H + NQ;              // [E,2] alpha -> ex (in place)
    float* AM  = AL + (size_t)EE * 2; // [N,2] segment max (uint keys, then float)
    float* DEN = AM + (size_t)NN * 2; // [N,2] softmax denom
    float* GS  = DEN + (size_t)NN * 2;// [G,128] pooled sums
    float* GC  = GS + (size_t)GG * DD;// [G] counts
    _Float16* PW = (_Float16*)(GC + GG); // packed f16 weights (4*PWCHUNK f16)

    for (int layer = 0; layer < 2; ++layer) {
        const int p = 4 + 9 * layer;
        const float* Wq  = (const float*)d_in[p + 0];
        const float* bq  = (const float*)d_in[p + 1];
        const float* Wk  = (const float*)d_in[p + 2];
        const float* bk  = (const float*)d_in[p + 3];
        const float* Wv  = (const float*)d_in[p + 4];
        const float* bv  = (const float*)d_in[p + 5];
        const float* We  = (const float*)d_in[p + 6];
        const float* Wsk = (const float*)d_in[p + 7];
        const float* bsk = (const float*)d_in[p + 8];
        const float* Xin = layer ? H : x;

        hipMemsetAsync(AM,  0, (size_t)NN * 2 * sizeof(float), stream);
        hipMemsetAsync(DEN, 0, (size_t)NN * 2 * sizeof(float), stream);

        pack_w<<<4, 256, 0, stream>>>(Wq,  PW);
        pack_w<<<4, 256, 0, stream>>>(Wk,  PW + PWCHUNK);
        pack_w<<<4, 256, 0, stream>>>(Wv,  PW + 2 * PWCHUNK);
        pack_w<<<4, 256, 0, stream>>>(Wsk, PW + 3 * PWCHUNK);

        gemm_qkvs<<<NN / 16, 256, 0, stream>>>(Xin, PW, bq, bk, bv, bsk,
                                               Q, K, V, S);
        edge_alpha<<<EE / 8, 256, 0, stream>>>(ei, Q, K, ea, We, AL,
                                               (unsigned int*)AM);
        amax_fix<<<(NN * 2 + 255) / 256, 256, 0, stream>>>(AM);
        edge_exp<<<(EE * 2 + 255) / 256, 256, 0, stream>>>(ei, AL, AM, DEN);
        edge_msg<<<EE / 8, 256, 0, stream>>>(ei, V, ea, We, AL, DEN, S);
        copy_act<<<((int)NQ + 255) / 256, 256, 0, stream>>>(
            S, layer ? out : H, (int)NQ, layer ? 0 : 1);
    }

    hipMemsetAsync(GS, 0, ((size_t)GG * DD + GG) * sizeof(float), stream);
    pool_sum<<<((int)NQ + 255) / 256, 256, 0, stream>>>(out, batch, GS);
    pool_cnt<<<(NN + 255) / 256, 256, 0, stream>>>(batch, GC);
    graph_mlp<<<GG, 128, 0, stream>>>(GS, GC, W1, b1, W2, b2, W3, b3,
                                      out + (size_t)NN * DD);
}
